// VisionTransformer_90890097918320
// MI455X (gfx1250) — matrix operand order repaired and checked by reading, not compiled, not run on hardware
//
#include <hip/hip_runtime.h>
#include <math.h>

// ---------------------------------------------------------------------------
// Types for CDNA5 WMMA (wave32, 16x16x32 bf16 -> f32)
// ---------------------------------------------------------------------------
typedef __bf16 bf16;
typedef __attribute__((ext_vector_type(4)))  __bf16 v4bf;
typedef __attribute__((ext_vector_type(8)))  __bf16 v8bf;
typedef __attribute__((ext_vector_type(16))) __bf16 v16bf;
typedef __attribute__((ext_vector_type(4)))  float  v4f;
typedef __attribute__((ext_vector_type(8)))  float  v8f;

#define EPI_BIAS  1
#define EPI_GELU  2
#define EPI_RES   4
#define EPI_SCALE 8

// ---------------------------------------------------------------------------
// Generalized batched strided GEMM:
//   C[z][m,n] = epi( scale * sum_k A[z][m,k] * W[z][k,n] + bias[n] + Res[z][m,n] )
// z decomposed as (zo, zi) with zi = z % inner (e.g. batch x head).
// Tile: 128(M) x 64(N) per 128-thread block; each wave: 32x64 via 8 WMMAs.
// ---------------------------------------------------------------------------
struct GemmP {
  const float* A; long long sAm, sAk, bsAo, bsAi;
  const float* W; long long sWk, sWn, bsWo, bsWi;
  float*       C; long long ldc, bsCo, bsCi;
  const float* bias;
  const float* res; long long ldr, bsRo, bsRi;
  int M, N, K, inner, epi;
  float scale;
};

__global__ __launch_bounds__(128) void gemm_wmma_bf16(GemmP p) {
  __shared__ bf16 As[128][40];  // [m][k], 80B row pitch (16B-aligned chunks)
  __shared__ bf16 Bs[64][40];   // [n][k]

  const int z  = blockIdx.z;
  const int zo = z / p.inner;
  const int zi = z - zo * p.inner;
  const float* __restrict__ A = p.A + zo * p.bsAo + zi * p.bsAi;
  const float* __restrict__ W = p.W + zo * p.bsWo + zi * p.bsWi;
  float* __restrict__       C = p.C + zo * p.bsCo + zi * p.bsCi;

  const int m0 = blockIdx.y * 128;
  const int n0 = blockIdx.x * 64;
  const int tid  = threadIdx.x;
  const int lane = tid & 31;
  const int wave = tid >> 5;
  const int half = lane >> 4;
  const int l16  = lane & 15;

  const bool fullN = (n0 + 64 <= p.N);
  const bool aVec  = (p.sAk == 1) && ((p.sAm & 3) == 0);           // row-clamped
  const bool bVecK = (p.sWk == 1) && ((p.sWn & 3) == 0);           // row-clamped
  const bool bVecN = fullN && (p.sWn == 1) && ((p.sWk & 3) == 0);  // weights

  v8f acc[2][4];
#pragma unroll
  for (int t = 0; t < 2; ++t)
#pragma unroll
    for (int s = 0; s < 4; ++s) acc[t][s] = (v8f)0.0f;

  for (int k0 = 0; k0 < p.K; k0 += 32) {
    const bool fullK = (k0 + 32 <= p.K);

    // ================= stage A tile (128 m x 32 k) =================
    if (aVec && fullK) {
#pragma unroll
      for (int i = 0; i < 8; ++i) {
        int idx = tid + i * 128;          // 1024 float4 slots
        int mi = idx >> 3;                // 128 rows
        int k4 = (idx & 7) * 4;
        int m  = m0 + mi;
        int mc = m < p.M ? m : (p.M - 1);
        float msk = (m < p.M) ? 1.0f : 0.0f;
        const float* src = A + (long long)mc * p.sAm + (k0 + k4);
        if (i == 0 && k0 + 32 < p.K) __builtin_prefetch(src + 32, 0, 0);
        v4f v = *(const v4f*)src;
        v4bf pk = { (bf16)(v[0] * msk), (bf16)(v[1] * msk),
                    (bf16)(v[2] * msk), (bf16)(v[3] * msk) };
        *(v4bf*)&As[mi][k4] = pk;
      }
    } else {
      // branch-free masked scalar path (K-tails / strided A)
#pragma unroll
      for (int i = 0; i < 32; ++i) {
        int idx = tid + i * 128;
        int mi = idx >> 5, ki = idx & 31;
        int m = m0 + mi, k = k0 + ki;
        int mc = m < p.M ? m : (p.M - 1);
        int kc = k < p.K ? k : (p.K - 1);
        float msk = (float)((m < p.M) & (k < p.K));
        float v = A[(long long)mc * p.sAm + (long long)kc * p.sAk] * msk;
        As[mi][ki] = (bf16)v;
      }
    }

    // ================= stage W tile as Bs[n][k] (64 n x 32 k) =================
    if (bVecK && fullK) {
      // contiguous along k (attention K^T-style operands), row-clamped
#pragma unroll
      for (int i = 0; i < 4; ++i) {
        int idx = tid + i * 128;
        int ni = idx >> 3;
        int k4 = (idx & 7) * 4;
        int n  = n0 + ni;
        int nc = n < p.N ? n : (p.N - 1);
        float msk = (n < p.N) ? 1.0f : 0.0f;
        const float* src = W + (long long)nc * p.sWn + (k0 + k4);
        if (i == 0 && k0 + 32 < p.K) __builtin_prefetch(src + 32, 0, 0);
        v4f v = *(const v4f*)src;
        v4bf pk = { (bf16)(v[0] * msk), (bf16)(v[1] * msk),
                    (bf16)(v[2] * msk), (bf16)(v[3] * msk) };
        *(v4bf*)&Bs[ni][k4] = pk;
      }
    } else if (bVecN && fullK) {
      // contiguous along n (weight matrices, K x N row-major)
#pragma unroll
      for (int i = 0; i < 4; ++i) {
        int idx = tid + i * 128;
        int ki = idx >> 4;                // 32 k
        int n4 = (idx & 15) * 4;          // 64 n
        const float* src = W + (long long)(k0 + ki) * p.sWk + (n0 + n4);
        if (i == 0 && k0 + 32 < p.K) __builtin_prefetch(src + 32 * p.sWk, 0, 0);
        v4f v = *(const v4f*)src;
        Bs[n4 + 0][ki] = (bf16)v[0];
        Bs[n4 + 1][ki] = (bf16)v[1];
        Bs[n4 + 2][ki] = (bf16)v[2];
        Bs[n4 + 3][ki] = (bf16)v[3];
      }
    } else {
#pragma unroll
      for (int i = 0; i < 16; ++i) {
        int idx = tid + i * 128;
        int ni = idx & 63, ki = idx >> 6;
        int n = n0 + ni, k = k0 + ki;
        int nc = n < p.N ? n : (p.N - 1);
        int kc = k < p.K ? k : (p.K - 1);
        float msk = (float)((n < p.N) & (k < p.K));
        float v = W[(long long)kc * p.sWk + (long long)nc * p.sWn] * msk;
        Bs[ni][ki] = (bf16)v;
      }
    }
    __syncthreads();

    // ---- A fragments (2 x 16x32 per wave), ISA VGPR layout:
    // lanes 0-15: K 0..7 & 16..23 ; lanes 16-31: K 8..15 & 24..31
    v16bf af[2];
#pragma unroll
    for (int t = 0; t < 2; ++t) {
      const v8bf* pa0 = (const v8bf*)&As[32 * wave + 16 * t + l16][half * 8];
      const v8bf* pa1 = (const v8bf*)&As[32 * wave + 16 * t + l16][16 + half * 8];
      v8bf a0 = *pa0, a1 = *pa1;
#pragma unroll
      for (int e = 0; e < 8; ++e) { af[t][e] = a0[e]; af[t][8 + e] = a1[e]; }
    }

    // ---- 4 B fragments, each reused by both A fragments: 8 WMMAs per K-step
#pragma unroll
    for (int s = 0; s < 4; ++s) {
      const v8bf* pb0 = (const v8bf*)&Bs[s * 16 + l16][half * 8];
      const v8bf* pb1 = (const v8bf*)&Bs[s * 16 + l16][16 + half * 8];
      v8bf b0 = *pb0, b1 = *pb1;
      v16bf bv;
#pragma unroll
      for (int e = 0; e < 8; ++e) { bv[e] = b0[e]; bv[8 + e] = b1[e]; }
      acc[0][s] = __builtin_amdgcn_wmma_f32_16x16x32_bf16(
          false, af[0], false, bv, (short)0, acc[0][s], false, false);
      acc[1][s] = __builtin_amdgcn_wmma_f32_16x16x32_bf16(
          false, af[1], false, bv, (short)0, acc[1][s], false, false);
    }
    __syncthreads();
  }

  // ---- epilogue: C layout VGPR r -> M = r + 8*half, N = lane&15
  const float* R = (p.epi & EPI_RES) ? (p.res + zo * p.bsRo + zi * p.bsRi) : nullptr;
#pragma unroll
  for (int t = 0; t < 2; ++t) {
#pragma unroll
    for (int s = 0; s < 4; ++s) {
      int n = n0 + s * 16 + l16;
      if (n < p.N) {
        float bia = (p.epi & EPI_BIAS) ? p.bias[n] : 0.0f;
#pragma unroll
        for (int r = 0; r < 8; ++r) {
          int m = m0 + 32 * wave + 16 * t + 8 * half + r;
          if (m < p.M) {
            float v = acc[t][s][r];
            if (p.epi & EPI_SCALE) v *= p.scale;
            v += bia;
            if (p.epi & EPI_GELU) v = 0.5f * v * (1.0f + erff(v * 0.70710678118654752f));
            if (p.epi & EPI_RES)  v += R[(long long)m * p.ldr + n];
            C[(long long)m * p.ldc + n] = v;
          }
        }
      }
    }
  }
}

// ---------------------------------------------------------------------------
// Reductions (wave32)
// ---------------------------------------------------------------------------
__device__ __forceinline__ float waveSum(float v) {
#pragma unroll
  for (int o = 16; o; o >>= 1) v += __shfl_xor(v, o, 32);
  return v;
}
__device__ __forceinline__ float waveMax(float v) {
#pragma unroll
  for (int o = 16; o; o >>= 1) v = fmaxf(v, __shfl_xor(v, o, 32));
  return v;
}
__device__ __forceinline__ float blockSum(float v, float* sm) {
  int lane = threadIdx.x & 31, w = threadIdx.x >> 5, nw = blockDim.x >> 5;
  v = waveSum(v);
  __syncthreads();
  if (!lane) sm[w] = v;
  __syncthreads();
  if (threadIdx.x == 0) { float t = sm[0]; for (int i = 1; i < nw; ++i) t += sm[i]; sm[0] = t; }
  __syncthreads();
  return sm[0];
}
__device__ __forceinline__ float blockMax(float v, float* sm) {
  int lane = threadIdx.x & 31, w = threadIdx.x >> 5, nw = blockDim.x >> 5;
  v = waveMax(v);
  __syncthreads();
  if (!lane) sm[w] = v;
  __syncthreads();
  if (threadIdx.x == 0) { float t = sm[0]; for (int i = 1; i < nw; ++i) t = fmaxf(t, sm[i]); sm[0] = t; }
  __syncthreads();
  return sm[0];
}

// ---------------------------------------------------------------------------
// LayerNorm over contiguous rows of length L
// ---------------------------------------------------------------------------
__global__ __launch_bounds__(256) void layernorm_k(const float* __restrict__ x,
                                                   const float* __restrict__ g,
                                                   const float* __restrict__ b,
                                                   float* __restrict__ y, int L) {
  __shared__ float sm[8];
  const float* xr = x + (long long)blockIdx.x * L;
  float*       yr = y + (long long)blockIdx.x * L;
  float invL = 1.0f / (float)L;
  float s = 0.0f;
  for (int i = threadIdx.x; i < L; i += blockDim.x) s += xr[i];
  float mean = blockSum(s, sm) * invL;
  float vs = 0.0f;
  for (int i = threadIdx.x; i < L; i += blockDim.x) { float d = xr[i] - mean; vs += d * d; }
  float rstd = rsqrtf(blockSum(vs, sm) * invL + 1e-5f);
  for (int i = threadIdx.x; i < L; i += blockDim.x)
    yr[i] = (xr[i] - mean) * rstd * g[i] + b[i];
}

// In-place row softmax (encoder scores), row length L
__global__ __launch_bounds__(128) void softmax_k(float* __restrict__ x, int L) {
  __shared__ float sm[4];
  float* xr = x + (long long)blockIdx.x * L;
  float mx = -3.4e38f;
  for (int i = threadIdx.x; i < L; i += blockDim.x) mx = fmaxf(mx, xr[i]);
  mx = blockMax(mx, sm);
  float s = 0.0f;
  for (int i = threadIdx.x; i < L; i += blockDim.x) { float e = __expf(xr[i] - mx); xr[i] = e; s += e; }
  float inv = 1.0f / blockSum(s, sm);
  for (int i = threadIdx.x; i < L; i += blockDim.x) xr[i] *= inv;
}

// Decoder softmax: read (b,q,h) rows of attn^T layout, write (b,h,q) rows. L=256.
__global__ __launch_bounds__(128) void softmax_perm_k(const float* __restrict__ in,
                                                      float* __restrict__ out) {
  __shared__ float sm[4];
  int r = blockIdx.x;
  int b = r / 144, rem = r - b * 144;
  int q = rem / 12, h = rem - q * 12;
  const float* xr = in  + ((long long)b * 144 + q * 12 + h) * 256;
  float*       yr = out + ((long long)b * 144 + h * 12 + q) * 256;
  float v0 = xr[threadIdx.x], v1 = xr[threadIdx.x + 128];
  float mx = blockMax(fmaxf(v0, v1), sm);
  float e0 = __expf(v0 - mx), e1 = __expf(v1 - mx);
  float inv = 1.0f / blockSum(e0 + e1, sm);
  yr[threadIdx.x]       = e0 * inv;
  yr[threadIdx.x + 128] = e1 * inv;
}

// ---------------------------------------------------------------------------
// Small elementwise kernels
// ---------------------------------------------------------------------------
__global__ void scale_k(const float* __restrict__ x, float* __restrict__ y, float s, int n) {
  int i = blockIdx.x * blockDim.x + threadIdx.x;
  if (i < n) y[i] = s * x[i];
}
__global__ void fill_cls_k(const float* __restrict__ cls_token,
                           const float* __restrict__ pos, float* __restrict__ H) {
  int i = blockIdx.x * blockDim.x + threadIdx.x;  // 8*768
  if (i < 8 * 768) {
    int b = i / 768, j = i - b * 768;
    H[(long long)b * 197376 + j] = cls_token[j] + pos[j];
  }
}
__global__ void copy_feat_k(const float* __restrict__ Y, float* __restrict__ F0,
                            float* __restrict__ out) {
  int i = blockIdx.x * blockDim.x + threadIdx.x;  // 8*256*768
  if (i < 8 * 256 * 768) {
    int b = i / (256 * 768), rem = i - b * (256 * 768);
    float v = Y[(long long)b * 197376 + 768 + rem];
    F0[i] = v; out[i] = v;
  }
}
__global__ void bcast_query_k(const float* __restrict__ cls_embed, float* __restrict__ Q) {
  int i = blockIdx.x * blockDim.x + threadIdx.x;  // 8*9216
  if (i < 8 * 9216) Q[i] = cls_embed[i % 9216];
}

// ---------------------------------------------------------------------------
// Host-side launcher
// ---------------------------------------------------------------------------
static inline void GEMM(hipStream_t st,
    const float* A, long long sAm, long long sAk, long long bsAo, long long bsAi,
    const float* W, long long sWk, long long sWn, long long bsWo, long long bsWi,
    float* C, long long ldc, long long bsCo, long long bsCi,
    int M, int N, int K, int batches, int inner, int epi,
    const float* bias, const float* res, long long ldr, long long bsRo, long long bsRi,
    float scale) {
  GemmP p;
  p.A = A; p.sAm = sAm; p.sAk = sAk; p.bsAo = bsAo; p.bsAi = bsAi;
  p.W = W; p.sWk = sWk; p.sWn = sWn; p.bsWo = bsWo; p.bsWi = bsWi;
  p.C = C; p.ldc = ldc; p.bsCo = bsCo; p.bsCi = bsCi;
  p.bias = bias; p.res = res; p.ldr = ldr; p.bsRo = bsRo; p.bsRi = bsRi;
  p.M = M; p.N = N; p.K = K; p.inner = inner; p.epi = epi; p.scale = scale;
  dim3 grid((N + 63) / 64, (M + 127) / 128, batches);
  hipLaunchKernelGGL(gemm_wmma_bf16, grid, dim3(128), 0, st, p);
}

extern "C" void kernel_launch(void* const* d_in, const int* in_sizes, int n_in,
                              void* d_out, int out_size, void* d_ws, size_t ws_size,
                              hipStream_t stream) {
  (void)in_sizes; (void)n_in; (void)out_size; (void)ws_size;
  const int B = 8, D = 768, DEPTH = 12, HEADS = 12, HD = 64;
  const int NCLASS = 12, NPATCH = 256, T = 257, HID = 3072;
  const int TOK = B * T;            // 2056
  const int FTOK = B * NPATCH;      // 2048
  const float SCL = 0.125f;         // HD^-0.5
  const long long FEATSZ = (long long)B * NPATCH * D;              // 1,572,864
  const long long ALS    = (long long)B * NCLASS * HEADS * NPATCH; // 294,912

  // --- inputs (setup_inputs dict order, leaves flattened) ---
  const float* x        = (const float*)d_in[0];
  const float* cls_tok  = (const float*)d_in[1];
  const float* pos      = (const float*)d_in[2];
  const float* cls_emb  = (const float*)d_in[3];
  const float* hyb_W    = (const float*)d_in[4];
  const float* hyb_b    = (const float*)d_in[5];
  const float* e_ln1g   = (const float*)d_in[6];
  const float* e_ln1b   = (const float*)d_in[7];
  const float* e_Wqkv   = (const float*)d_in[8];
  const float* e_Wp     = (const float*)d_in[9];
  const float* e_bp     = (const float*)d_in[10];
  const float* e_ln2g   = (const float*)d_in[11];
  const float* e_ln2b   = (const float*)d_in[12];
  const float* e_W1     = (const float*)d_in[13];
  const float* e_b1     = (const float*)d_in[14];
  const float* e_W2     = (const float*)d_in[15];
  const float* e_b2     = (const float*)d_in[16];
  const float* d_ln1g   = (const float*)d_in[17];
  const float* d_ln1b   = (const float*)d_in[18];
  const float* d_ln1cg  = (const float*)d_in[19];
  const float* d_ln1cb  = (const float*)d_in[20];
  const float* d_Wq     = (const float*)d_in[21];
  const float* d_Wkv    = (const float*)d_in[22];
  const float* d_Wp     = (const float*)d_in[23];
  const float* d_bp     = (const float*)d_in[24];
  const float* d_ln2g   = (const float*)d_in[25];
  const float* d_ln2b   = (const float*)d_in[26];
  const float* m1_W1    = (const float*)d_in[27];
  const float* m1_b1    = (const float*)d_in[28];
  const float* m1_W2    = (const float*)d_in[29];
  const float* m1_b2    = (const float*)d_in[30];
  const float* d_ln3g   = (const float*)d_in[31];
  const float* d_ln3b   = (const float*)d_in[32];
  const float* m2_W1    = (const float*)d_in[33];
  const float* m2_b1    = (const float*)d_in[34];
  const float* m2_W2    = (const float*)d_in[35];
  const float* m2_b2    = (const float*)d_in[36];
  const float* d_ln4g   = (const float*)d_in[37];
  const float* d_ln4b   = (const float*)d_in[38];
  const float* m3_W1    = (const float*)d_in[39];
  const float* m3_b1    = (const float*)d_in[40];
  const float* m3_W2    = (const float*)d_in[41];
  const float* m3_b2    = (const float*)d_in[42];
  const float* norm_g   = (const float*)d_in[43];
  const float* norm_b   = (const float*)d_in[44];
  float* out = (float*)d_out;

  // --- workspace carve ---
  float* ws = (float*)d_ws;
  size_t o = 0;
  auto take = [&](size_t n) { float* p = ws + o; o += n; return p; };
  float* H0  = take((size_t)TOK * D);
  float* H1  = take((size_t)TOK * D);
  float* Y   = take((size_t)TOK * D);
  float* QKV = take((size_t)TOK * 3 * D);     // also decoder KV (2048x1536)
  float* S   = take((size_t)96 * T * T);      // encoder scores
  float* OB  = take((size_t)TOK * D);
  float* MLPH= take((size_t)TOK * HID);
  float* F0  = take((size_t)FTOK * D);
  float* F1  = take((size_t)FTOK * D);
  float* Q0  = take((size_t)96 * D);
  float* Q1  = take((size_t)96 * D);
  float* QN  = take((size_t)96 * D);
  float* QP  = take((size_t)96 * D);
  float* AR  = take((size_t)ALS);             // raw attn (b,q,h,k) layout
  float* AS  = take((size_t)ALS);             // softmaxed (b,h,q,k) / LN buf
  float* A2  = take((size_t)ALS);             // 2*raw attn

  // =========================== patch embed =================================
  // h[b,p,:] = x[b,:,p]^T @ hybrid_W + hybrid_b + pos[p+1]  (rows 1..256)
  GEMM(stream, x, /*sAm*/1, /*sAk*/256, /*bsAo*/2048LL * 256, 0,
       hyb_W, D, 1, 0, 0,
       H0 + D, D, (long long)T * D, 0,
       NPATCH, D, 2048, B, 1, EPI_BIAS | EPI_RES,
       hyb_b, pos + D, D, 0, 0, 1.0f);
  fill_cls_k<<<(B * D + 255) / 256, 256, 0, stream>>>(cls_tok, pos, H0);

  // =========================== encoder =====================================
  for (int l = 0; l < DEPTH; ++l) {
    float* Hc = (l & 1) ? H1 : H0;
    float* Hn = (l & 1) ? H0 : H1;
    const float* Wqkv = e_Wqkv + (long long)l * D * 3 * D;
    const float* Wp   = e_Wp   + (long long)l * D * D;
    const float* W1   = e_W1   + (long long)l * D * HID;
    const float* W2   = e_W2   + (long long)l * HID * D;

    layernorm_k<<<TOK, 256, 0, stream>>>(Hc, e_ln1g + l * D, e_ln1b + l * D, Y, D);
    // qkv = y @ Wqkv
    GEMM(stream, Y, D, 1, 0, 0, Wqkv, 3 * D, 1, 0, 0,
         QKV, 3 * D, 0, 0, TOK, 3 * D, D, 1, 1, 0,
         nullptr, nullptr, 0, 0, 0, 1.0f);
    // S[b,h] = Q @ K^T * scale
    GEMM(stream, QKV, 3 * D, 1, (long long)T * 3 * D, HD,
         QKV + D, 1, 3 * D, (long long)T * 3 * D, HD,
         S, T, (long long)HEADS * T * T, (long long)T * T,
         T, T, HD, B * HEADS, HEADS, EPI_SCALE,
         nullptr, nullptr, 0, 0, 0, SCL);
    softmax_k<<<96 * T, 128, 0, stream>>>(S, T);
    // O[b,h] = A @ V
    GEMM(stream, S, T, 1, (long long)HEADS * T * T, (long long)T * T,
         QKV + 2 * D, 3 * D, 1, (long long)T * 3 * D, HD,
         OB, D, (long long)T * D, HD,
         T, HD, T, B * HEADS, HEADS, 0,
         nullptr, nullptr, 0, 0, 0, 1.0f);
    // h = h + O @ Wp + bp
    GEMM(stream, OB, D, 1, 0, 0, Wp, D, 1, 0, 0,
         Hn, D, 0, 0, TOK, D, D, 1, 1, EPI_BIAS | EPI_RES,
         e_bp + l * D, Hc, D, 0, 0, 1.0f);
    layernorm_k<<<TOK, 256, 0, stream>>>(Hn, e_ln2g + l * D, e_ln2b + l * D, Y, D);
    GEMM(stream, Y, D, 1, 0, 0, W1, HID, 1, 0, 0,
         MLPH, HID, 0, 0, TOK, HID, D, 1, 1, EPI_BIAS | EPI_GELU,
         e_b1 + l * HID, nullptr, 0, 0, 0, 1.0f);
    GEMM(stream, MLPH, HID, 1, 0, 0, W2, D, 1, 0, 0,
         Hn, D, 0, 0, TOK, D, HID, 1, 1, EPI_BIAS | EPI_RES,
         e_b2 + l * D, Hn, D, 0, 0, 1.0f);
  }
  // final LN + feat
  layernorm_k<<<TOK, 256, 0, stream>>>(H0, norm_g, norm_b, Y, D);
  copy_feat_k<<<(int)((FEATSZ + 255) / 256), 256, 0, stream>>>(Y, F0, out);
  bcast_query_k<<<(96 * D + 255) / 256, 256, 0, stream>>>(cls_emb, Q0);

  // =========================== decoder =====================================
  for (int l = 0; l < DEPTH; ++l) {
    float* Qc = (l & 1) ? Q1 : Q0;
    float* Qn = (l & 1) ? Q0 : Q1;
    float* Fc = (l & 1) ? F1 : F0;
    float* Fn = (l & 1) ? F0 : F1;
    const float* Wq  = d_Wq  + (long long)l * D * D;
    const float* Wkv = d_Wkv + (long long)l * D * 2 * D;
    const float* Wp  = d_Wp  + (long long)l * D * D;
    float* KV = QKV;  // reuse

    layernorm_k<<<96, 256, 0, stream>>>(Qc, d_ln1cg + l * D, d_ln1cb + l * D, QN, D);
    layernorm_k<<<FTOK, 256, 0, stream>>>(Fc, d_ln1g + l * D, d_ln1b + l * D, Y, D);
    GEMM(stream, QN, D, 1, 0, 0, Wq, D, 1, 0, 0,
         QP, D, 0, 0, 96, D, D, 1, 1, 0, nullptr, nullptr, 0, 0, 0, 1.0f);
    GEMM(stream, Y, D, 1, 0, 0, Wkv, 2 * D, 1, 0, 0,
         KV, 2 * D, 0, 0, FTOK, 2 * D, D, 1, 1, 0, nullptr, nullptr, 0, 0, 0, 1.0f);
    // attn1[b,h,q,k] stored transposed as AR[b,q,h,k]; reference reshape quirk:
    // q[b,h,qc,d] = qproj[b, token=h, qc*64+d]
    GEMM(stream, QP, HD, 1, (long long)NCLASS * D, D,
         KV, 1, 2 * D, (long long)NPATCH * 2 * D, HD,
         AR, (long long)HEADS * NPATCH, (long long)NCLASS * HEADS * NPATCH, NPATCH,
         NCLASS, NPATCH, HD, B * HEADS, HEADS, EPI_SCALE,
         nullptr, nullptr, 0, 0, 0, SCL);
    softmax_perm_k<<<B * 144, 128, 0, stream>>>(AR, AS);
    // o[b,h,q,d] -> OB[b, h, q*64+d]
    GEMM(stream, AS, NPATCH, 1, (long long)NCLASS * HEADS * NPATCH, (long long)NCLASS * NPATCH,
         KV + D, 2 * D, 1, (long long)NPATCH * 2 * D, HD,
         OB, HD, (long long)NCLASS * D, D,
         NCLASS, HD, NPATCH, B * HEADS, HEADS, 0,
         nullptr, nullptr, 0, 0, 0, 1.0f);
    // query = o @ Wp + bp ; then *=2 ; then += mlp(LN(query))
    GEMM(stream, OB, D, 1, 0, 0, Wp, D, 1, 0, 0,
         Qn, D, 0, 0, 96, D, D, 1, 1, EPI_BIAS,
         d_bp + l * D, nullptr, 0, 0, 0, 1.0f);
    scale_k<<<(96 * D + 255) / 256, 256, 0, stream>>>(Qn, Qn, 2.0f, 96 * D);
    layernorm_k<<<96, 256, 0, stream>>>(Qn, d_ln2g + l * D, d_ln2b + l * D, QN, D);
    GEMM(stream, QN, D, 1, 0, 0, m1_W1 + (long long)l * D * HID, HID, 1, 0, 0,
         MLPH, HID, 0, 0, 96, HID, D, 1, 1, EPI_BIAS | EPI_GELU,
         m1_b1 + l * HID, nullptr, 0, 0, 0, 1.0f);
    GEMM(stream, MLPH, HID, 1, 0, 0, m1_W2 + (long long)l * HID * D, D, 1, 0, 0,
         Qn, D, 0, 0, 96, D, HID, 1, 1, EPI_BIAS | EPI_RES,
         m1_b2 + l * D, Qn, D, 0, 0, 1.0f);
    // f = 2f ; f += mlp(LN(f))
    scale_k<<<(int)((FEATSZ + 255) / 256), 256, 0, stream>>>(Fc, Fn, 2.0f, (int)FEATSZ);
    layernorm_k<<<FTOK, 256, 0, stream>>>(Fn, d_ln3g + l * D, d_ln3b + l * D, Y, D);
    GEMM(stream, Y, D, 1, 0, 0, m2_W1 + (long long)l * D * HID, HID, 1, 0, 0,
         MLPH, HID, 0, 0, FTOK, HID, D, 1, 1, EPI_BIAS | EPI_GELU,
         m2_b1 + l * HID, nullptr, 0, 0, 0, 1.0f);
    GEMM(stream, MLPH, HID, 1, 0, 0, m2_W2 + (long long)l * HID * D, D, 1, 0, 0,
         Fn, D, 0, 0, FTOK, D, HID, 1, 1, EPI_BIAS | EPI_RES,
         m2_b2 + l * D, Fn, D, 0, 0, 1.0f);
    // attn = 2*attn1^T ; attn += mlp(LN(attn)) over NPATCH ; write to output
    scale_k<<<(int)((ALS + 255) / 256), 256, 0, stream>>>(AR, A2, 2.0f, (int)ALS);
    layernorm_k<<<B * 144, 256, 0, stream>>>(A2, d_ln4g + l * NPATCH, d_ln4b + l * NPATCH, AS, NPATCH);
    GEMM(stream, AS, NPATCH, 1, 0, 0, m3_W1 + (long long)l * NPATCH * 3 * NPATCH, 3 * NPATCH, 1, 0, 0,
         MLPH, 3 * NPATCH, 0, 0, B * 144, 3 * NPATCH, NPATCH, 1, 1, EPI_BIAS | EPI_GELU,
         m3_b1 + l * 3 * NPATCH, nullptr, 0, 0, 0, 1.0f);
    GEMM(stream, MLPH, 3 * NPATCH, 1, 0, 0, m3_W2 + (long long)l * 3 * NPATCH * NPATCH, NPATCH, 1, 0, 0,
         out + FEATSZ + (long long)l * ALS, NPATCH, 0, 0,
         B * 144, NPATCH, 3 * NPATCH, 1, 1, EPI_BIAS | EPI_RES,
         m3_b2 + l * NPATCH, A2, NPATCH, 0, 0, 1.0f);
  }
}
